// GAT_78005196030462
// MI455X (gfx1250) — matrix operand order repaired and checked by reading, not compiled, not run on hardware
//
#include <hip/hip_runtime.h>
#include <hip/hip_bf16.h>
#include <math.h>

typedef __attribute__((ext_vector_type(16))) __bf16 bf16x16;
typedef __attribute__((ext_vector_type(8)))  float  f32x8;

#define NEG_SLOPE 0.2f

// ---------------------------------------------------------------------------
// WMMA GEMM: C[M,Nc] = A[M,K] @ W[K,Nc]; fp32 memory, bf16 WMMA, fp32 acc.
// Each wave computes FIVE 16x16 m-tiles against one 16-col n-tile: the B
// fragment is loaded once per K-step and reused by 5 straight-line WMMAs.
// mTiles (=M/16) MUST be divisible by 5 (50000/16 = 3125 = 5^5), so there
// are NO per-subtile guards -> no divergent exec masking around WMMA.
// blockDim = 256 (8 waves). Only guard is a whole-wave early return.
// ---------------------------------------------------------------------------
#define MT_PER_WAVE 5

__global__ void __launch_bounds__(256)
gat_wmma_gemm(const float* __restrict__ A, const float* __restrict__ W,
              float* __restrict__ C, int M, int K, int Nc) {
    const int lane   = threadIdx.x & 31;
    const int wave   = threadIdx.x >> 5;
    const int mTiles = M >> 4;
    const int mtile0 = (blockIdx.x * 8 + wave) * MT_PER_WAVE;
    if (mtile0 >= mTiles) return;         // whole-wave exit; EXEC all-1s below

    const int n0    = blockIdx.y << 4;
    const int mrow  = lane & 15;
    const int khalf = lane >> 4;          // 0 or 1
    const int ncol  = n0 + (lane & 15);

    f32x8 acc[MT_PER_WAVE] = {};

    for (int kk = 0; kk < K; kk += 32) {
        // ---- B fragment (32x16 bf16): col = lane&15, K = khalf*16 + e ----
        bf16x16 b;
        const float* Wp = W + (long long)(kk + khalf * 16) * Nc + ncol;
#pragma unroll
        for (int e = 0; e < 16; ++e)
            b[e] = (__bf16)Wp[(long long)e * Nc];

        // ---- 5 A fragments reuse b, back-to-back WMMAs ----
#pragma unroll
        for (int t = 0; t < MT_PER_WAVE; ++t) {
            bf16x16 a;
            const float* Arow =
                A + (long long)(((mtile0 + t) << 4) + mrow) * K + kk;
#pragma unroll
            for (int v = 0; v < 8; ++v) {
                const int kb = khalf * 16 + v * 2;
                float2 av = *reinterpret_cast<const float2*>(Arow + kb);
                a[2 * v]     = (__bf16)av.x;
                a[2 * v + 1] = (__bf16)av.y;
            }
            acc[t] = __builtin_amdgcn_wmma_f32_16x16x32_bf16(
                false, a, false, b, (short)0, acc[t], false, false);
        }
    }

    // ---- store: VGPR r -> row r + khalf*8, col = lane&15 ----
#pragma unroll
    for (int t = 0; t < MT_PER_WAVE; ++t) {
#pragma unroll
        for (int r = 0; r < 8; ++r) {
            const int row = ((mtile0 + t) << 4) + r + khalf * 8;
            C[(long long)row * Nc + ncol] = acc[t][r];
        }
    }
}

// ---------------------------------------------------------------------------
// Utility fill
// ---------------------------------------------------------------------------
__global__ void fill_kernel(float* p, float v, long long n) {
    long long i = (long long)blockIdx.x * blockDim.x + threadIdx.x;
    if (i < n) p[i] = v;
}

// ---------------------------------------------------------------------------
// Per-node attention scalars, one wave per node. Lanes read contiguous
// channels (coalesced), cross-lane reduce via shfl_xor (wave32).
// a_s[n,h] = sum_c h[n,h*C+c]*att_src[h,c]
// ---------------------------------------------------------------------------
__global__ void __launch_bounds__(256)
attn_kernel(const float* __restrict__ h,
            const float* __restrict__ att_s,
            const float* __restrict__ att_d,
            float* __restrict__ out_s, float* __restrict__ out_d,
            long long N, int H, int C) {
    const long long n = (long long)blockIdx.x * (blockDim.x >> 5) + (threadIdx.x >> 5);
    if (n >= N) return;
    const int lane = threadIdx.x & 31;
    const float* hp = h + n * (long long)(H * C);
    for (int hh = 0; hh < H; ++hh) {
        float ss = 0.f, dd = 0.f;
        for (int c = lane; c < C; c += 32) {
            const float v = hp[hh * C + c];
            ss += v * att_s[hh * C + c];
            dd += v * att_d[hh * C + c];
        }
#pragma unroll
        for (int off = 16; off > 0; off >>= 1) {
            ss += __shfl_xor(ss, off, 32);
            dd += __shfl_xor(dd, off, 32);
        }
        if (lane == 0) {
            out_s[n * H + hh] = ss;
            out_d[n * H + hh] = dd;
        }
    }
}

// ---------------------------------------------------------------------------
// Float atomic max via int punning (correct for mixed signs; NaN skipped)
// ---------------------------------------------------------------------------
__device__ __forceinline__ void atomicMaxF(float* addr, float val) {
    if (!(val == val)) return;
    if (val >= 0.f)
        atomicMax((int*)addr, __float_as_int(val));
    else
        atomicMin((unsigned int*)addr, __float_as_uint(val));
}

__device__ __forceinline__ void edge_sd(const long long* ei, long long E,
                                        long long e, long long& s, long long& d) {
    if (e < E) { s = ei[e]; d = ei[E + e]; }
    else       { s = d = e - E; }          // appended self-loops
}

// Pass 1: segment max of leaky-relu'd logits over dst
__global__ void edge_max_kernel(const long long* __restrict__ ei, long long E,
                                long long Ep,
                                const float* __restrict__ a_s,
                                const float* __restrict__ a_d,
                                float* __restrict__ emax, int H) {
    long long e = (long long)blockIdx.x * blockDim.x + threadIdx.x;
    if (e >= Ep) return;
    long long s, d; edge_sd(ei, E, e, s, d);
    for (int h = 0; h < H; ++h) {
        float t = a_s[s * H + h] + a_d[d * H + h];
        t = t > 0.f ? t : NEG_SLOPE * t;
        atomicMaxF(&emax[d * H + h], t);
    }
}

// Pass 2: e_exp = exp(e - max); denom[dst] += e_exp
__global__ void edge_exp_kernel(const long long* __restrict__ ei, long long E,
                                long long Ep,
                                const float* __restrict__ a_s,
                                const float* __restrict__ a_d,
                                const float* __restrict__ emax,
                                float* __restrict__ eexp,
                                float* __restrict__ denom, int H) {
    long long e = (long long)blockIdx.x * blockDim.x + threadIdx.x;
    if (e >= Ep) return;
    long long s, d; edge_sd(ei, E, e, s, d);
    for (int h = 0; h < H; ++h) {
        float t = a_s[s * H + h] + a_d[d * H + h];
        t = t > 0.f ? t : NEG_SLOPE * t;
        float ex = __expf(t - emax[d * H + h]);
        eexp[e * H + h] = ex;
        atomicAdd(&denom[d * H + h], ex);
    }
}

// Pass 3: agg[dst, h, c] += h[src, h, c] * (e_exp / denom[dst, h]).
// One wave per edge; each lane owns HC/32 CONSECUTIVE channels so the src
// gather merges into b128/b64 loads. All buffers L2-resident on MI455X.
__global__ void __launch_bounds__(256)
edge_aggregate_kernel(const long long* __restrict__ ei, long long E,
                      long long Ep,
                      const float* __restrict__ hfeat,
                      const float* __restrict__ eexp,
                      const float* __restrict__ denom,
                      float* __restrict__ agg, int H, int C) {
    const long long e = (long long)blockIdx.x * (blockDim.x >> 5) + (threadIdx.x >> 5);
    if (e >= Ep) return;
    const int lane = threadIdx.x & 31;
    long long s, d; edge_sd(ei, E, e, s, d);
    const int HC  = H * C;
    const int per = HC >> 5;                       // channels per lane (4 or 2)
    const float* sp = hfeat + s * (long long)HC + lane * per;
    float*       dp = agg   + d * (long long)HC + lane * per;
#pragma unroll 4
    for (int j = 0; j < per; ++j) {
        const int h = (lane * per + j) / C;
        const float alpha = eexp[e * H + h] / denom[d * H + h];
        atomicAdd(&dp[j], sp[j] * alpha);
    }
}

// Bias add, optional ELU (alpha=1)
__global__ void bias_act_kernel(float* __restrict__ x, const float* __restrict__ bias,
                                long long N, int F, int do_elu) {
    long long i = (long long)blockIdx.x * blockDim.x + threadIdx.x;
    if (i >= N * F) return;
    const int j = (int)(i % F);
    float v = x[i] + bias[j];
    if (do_elu) v = v > 0.f ? v : (__expf(v) - 1.f);
    x[i] = v;
}

// ---------------------------------------------------------------------------
// Host orchestration
// ---------------------------------------------------------------------------
extern "C" void kernel_launch(void* const* d_in, const int* in_sizes, int n_in,
                              void* d_out, int out_size, void* d_ws, size_t ws_size,
                              hipStream_t stream) {
    const float*     x   = (const float*)d_in[0];
    const long long* ei  = (const long long*)d_in[1];   // int64 [2,E]
    const float*     W1  = (const float*)d_in[2];
    const float*     as1 = (const float*)d_in[3];
    const float*     ad1 = (const float*)d_in[4];
    const float*     b1  = (const float*)d_in[5];
    const float*     W2  = (const float*)d_in[6];
    const float*     as2 = (const float*)d_in[7];
    const float*     ad2 = (const float*)d_in[8];
    const float*     b2  = (const float*)d_in[9];
    float* out = (float*)d_out;

    const int  F_in = 128;
    const long long N  = in_sizes[0] / F_in;        // 50000
    const long long E  = in_sizes[1] / 2;           // 800000
    const long long Ep = E + N;                     // + self loops
    const int  HC1 = in_sizes[5];                   // 128
    const int  H1  = 4;                             // heads (reference)
    const int  C1  = HC1 / H1;                      // 32
    const int  F2  = in_sizes[9];                   // 64

    // Workspace layout (fp32)
    float* ws    = (float*)d_ws;
    float* h1    = ws;                    // N*HC1
    float* agg1  = h1    + N * HC1;       // N*HC1
    float* as1n  = agg1  + N * HC1;       // N*H1
    float* ad1n  = as1n  + N * H1;        // N*H1
    float* emax1 = ad1n  + N * H1;        // N*H1
    float* den1  = emax1 + N * H1;        // N*H1
    float* eexp1 = den1  + N * H1;        // Ep*H1
    float* h2    = eexp1 + Ep * H1;       // N*F2
    float* as2n  = h2    + N * F2;        // N
    float* ad2n  = as2n  + N;             // N
    float* emax2 = ad2n  + N;             // N
    float* den2  = emax2 + N;             // N
    float* eexp2 = den2  + N;             // Ep

    const float NINF = -INFINITY;
    auto blocks = [](long long n, int bs) { return (unsigned)((n + bs - 1) / bs); };

    // Init accumulators
    fill_kernel<<<blocks(N * HC1, 256), 256, 0, stream>>>(agg1, 0.f, N * HC1);
    fill_kernel<<<blocks(N * H1, 256), 256, 0, stream>>>(den1, 0.f, N * H1);
    fill_kernel<<<blocks(N * H1, 256), 256, 0, stream>>>(emax1, NINF, N * H1);
    fill_kernel<<<blocks(N * F2, 256), 256, 0, stream>>>(out, 0.f, N * F2);
    fill_kernel<<<blocks(N, 256), 256, 0, stream>>>(den2, 0.f, N);
    fill_kernel<<<blocks(N, 256), 256, 0, stream>>>(emax2, NINF, N);

    // ---- Layer 1 ----
    {
        const long long mGroups = (N >> 4) / MT_PER_WAVE;       // 3125/5 = 625
        dim3 grid(blocks(mGroups, 8), HC1 / 16);                // 8 waves/block
        gat_wmma_gemm<<<grid, 256, 0, stream>>>(x, W1, h1, (int)N, F_in, HC1);
    }
    attn_kernel<<<blocks(N, 8), 256, 0, stream>>>(h1, as1, ad1, as1n, ad1n, N, H1, C1);
    edge_max_kernel<<<blocks(Ep, 256), 256, 0, stream>>>(ei, E, Ep, as1n, ad1n, emax1, H1);
    edge_exp_kernel<<<blocks(Ep, 256), 256, 0, stream>>>(ei, E, Ep, as1n, ad1n, emax1, eexp1, den1, H1);
    edge_aggregate_kernel<<<blocks(Ep, 8), 256, 0, stream>>>(ei, E, Ep, h1, eexp1, den1, agg1, H1, C1);
    bias_act_kernel<<<blocks(N * HC1, 256), 256, 0, stream>>>(agg1, b1, N, HC1, 1);   // +b1, ELU

    // ---- Layer 2 ----
    {
        const long long mGroups = (N >> 4) / MT_PER_WAVE;
        dim3 grid(blocks(mGroups, 8), F2 / 16);
        gat_wmma_gemm<<<grid, 256, 0, stream>>>(agg1, W2, h2, (int)N, HC1, F2);
    }
    attn_kernel<<<blocks(N, 8), 256, 0, stream>>>(h2, as2, ad2, as2n, ad2n, N, 1, F2);
    edge_max_kernel<<<blocks(Ep, 256), 256, 0, stream>>>(ei, E, Ep, as2n, ad2n, emax2, 1);
    edge_exp_kernel<<<blocks(Ep, 256), 256, 0, stream>>>(ei, E, Ep, as2n, ad2n, emax2, eexp2, den2, 1);
    edge_aggregate_kernel<<<blocks(Ep, 8), 256, 0, stream>>>(ei, E, Ep, h2, eexp2, den2, out, 1, F2);
    bias_act_kernel<<<blocks(N * F2, 256), 256, 0, stream>>>(out, b2, N, F2, 0);      // +b2
}
